// Attention_80891414053176
// MI455X (gfx1250) — matrix operand order repaired and checked by reading, not compiled, not run on hardware
//
#include <hip/hip_runtime.h>
#include <hip/hip_bf16.h>
#include <stdint.h>

// MI455X / gfx1250, wave32. Fused masked-softmax attention:
//   scores = Q @ encT  (f16 WMMA, f32 acc), softmax w/ length mask, context = P @ enc
// Outputs: d_out = [context (8*1024*256 f32)] ++ [attn (8*1024*4096 f32)]

typedef _Float16 v16h __attribute__((ext_vector_type(16)));
typedef float    v8f  __attribute__((ext_vector_type(8)));

#define NB       8
#define OUT_LEN  1024
#define IN_LEN   4096
#define HDIM     256
#define M_TILE   64          // rows per workgroup (4 waves x 16 rows, x2 n-splits)
#define NSPLIT   2           // n-range splits (wave groups of 4 waves)
#define CHUNK    32          // enc rows staged per iteration
#define ASTRIDE  528         // bytes/enc-row, pass-A LDS layout [n][h] f16 (512+16 pad)
#define TSTRIDE  80          // bytes/h-row,  pass-B LDS layout [h][n] f16 (64+16 pad)
#define BUFBYTES 20480       // max(32*528, 256*80)
#define NEGBIG   (-1.0e30f)  // finite "masked" score: avoids inf/NaN in reductions

union Frag {
  v16h     v;
  unsigned u[8];
};

__device__ __forceinline__ unsigned pk2(float a, float b) {
  // v_cvt_pk_f16_f32-style pack; builtin returns __fp16 ext_vector_type(2)
  return __builtin_bit_cast(unsigned, __builtin_amdgcn_cvt_pkrtz(a, b));
}

__global__ __launch_bounds__(256) void
attn_fused_kernel(const float* __restrict__ q,     // [B, OUT_LEN, H]
                  const float* __restrict__ enc,   // [B, IN_LEN, H]
                  const int*   __restrict__ lens,  // [B]
                  float* __restrict__ ctx,         // [B, OUT_LEN, H]
                  float* __restrict__ attn)        // [B, OUT_LEN, IN_LEN]
{
  __shared__ __align__(16) char ldsBuf[NSPLIT * BUFBYTES];  // enc staging / C combine
  __shared__ float ldsMax2[M_TILE][NSPLIT];
  __shared__ float ldsSum2[M_TILE][NSPLIT];

  const int tid    = threadIdx.x;
  const int lane   = tid & 31;
  const int w      = tid >> 5;           // wave 0..7
  const int msub   = w & 3;              // m sub-tile (16 rows)
  const int nsplit = w >> 2;             // n half: [0,2048) or [2048,4096)
  const int l15    = lane & 15;
  const int hi16   = lane >> 4;          // lane half (WMMA layout selector)
  const int b      = (int)blockIdx.x >> 4;
  const int mt     = (int)blockIdx.x & 15;
  const int m0     = mt * M_TILE + msub * 16;
  const int len    = lens[b];
  const int gt     = tid & 127;          // thread id within nsplit group
  const int nBase  = nsplit * (IN_LEN / NSPLIT);

  const float* encB  = enc  + (size_t)b * IN_LEN * HDIM;
  float*       attnB = attn + (size_t)b * OUT_LEN * IN_LEN;
  char* buf = ldsBuf + nsplit * BUFBYTES;

  float rowmax;  // full-row max for row (m0 + l15), broadcast to all lanes

  // ======================= Pass A: scores + raw write + row max ==============
  {
    // Q tile: 16 rows x 256 h, resident as f16 A-fragments (ISA A-layout:
    // lanes<16 hold K {0..7,16..23}, lanes>=16 hold K {8..15,24..31} per 32-chunk)
    Frag aQ[8];
    {
      const float* qrow = q + ((size_t)(b * OUT_LEN + m0 + l15)) * HDIM + hi16 * 8;
      #pragma unroll
      for (int kc = 0; kc < 8; ++kc) {
        float4 x0 = *(const float4*)(qrow + kc * 32 + 0);
        float4 x1 = *(const float4*)(qrow + kc * 32 + 4);
        float4 y0 = *(const float4*)(qrow + kc * 32 + 16);
        float4 y1 = *(const float4*)(qrow + kc * 32 + 20);
        aQ[kc].u[0] = pk2(x0.x, x0.y); aQ[kc].u[1] = pk2(x0.z, x0.w);
        aQ[kc].u[2] = pk2(x1.x, x1.y); aQ[kc].u[3] = pk2(x1.z, x1.w);
        aQ[kc].u[4] = pk2(y0.x, y0.y); aQ[kc].u[5] = pk2(y0.z, y0.w);
        aQ[kc].u[6] = pk2(y1.x, y1.y); aQ[kc].u[7] = pk2(y1.z, y1.w);
      }
    }

    float vmax[8];
    #pragma unroll
    for (int r = 0; r < 8; ++r) vmax[r] = NEGBIG;

    #pragma unroll 1
    for (int c0 = 0; c0 < IN_LEN / NSPLIT; c0 += CHUNK) {
      const int n0 = nBase + c0;
      __syncthreads();
      // stage CHUNK x 256 f32 -> f16, [n][h] layout, padded stride (bank-safe)
      #pragma unroll
      for (int c = 0; c < 16; ++c) {
        const int row = c * 2 + (gt >> 6);
        const int hq  = gt & 63;
        float4 v = *(const float4*)(encB + (size_t)(n0 + row) * HDIM + hq * 4);
        uint2 p; p.x = pk2(v.x, v.y); p.y = pk2(v.z, v.w);
        *(uint2*)(buf + row * ASTRIDE + hq * 8) = p;
      }
      __syncthreads();

      #pragma unroll
      for (int t = 0; t < 2; ++t) {     // two 16-wide n tiles per chunk
        v8f acc = {0.f,0.f,0.f,0.f,0.f,0.f,0.f,0.f};
        const char* bb = buf + (t * 16 + l15) * ASTRIDE + hi16 * 16;
        #pragma unroll
        for (int kc = 0; kc < 8; ++kc) {  // K = h, 8 chunks of 32
          Frag bf;
          uint4 lo = *(const uint4*)(bb + kc * 64);
          uint4 hi = *(const uint4*)(bb + kc * 64 + 32);
          bf.u[0]=lo.x; bf.u[1]=lo.y; bf.u[2]=lo.z; bf.u[3]=lo.w;
          bf.u[4]=hi.x; bf.u[5]=hi.y; bf.u[6]=hi.z; bf.u[7]=hi.w;
          acc = __builtin_amdgcn_wmma_f32_16x16x32_f16(
                    false, aQ[kc].v, false, bf.v, (short)0, acc, false, false);
        }
        const int  col   = n0 + t * 16 + l15;
        const bool valid = col < len;
        float* arow = attnB + (size_t)(m0 + hi16 * 8) * IN_LEN + col;
        #pragma unroll
        for (int r = 0; r < 8; ++r) {   // D layout: VGPR r -> row r (+8 upper half)
          float s = valid ? acc[r] : NEGBIG;
          vmax[r] = fmaxf(vmax[r], s);
          arow[(size_t)r * IN_LEN] = s;  // raw masked scores -> final attn slot
        }
      }
    }

    // row max: reduce across the 16 column-slices (lanes of one half)
    #pragma unroll
    for (int r = 0; r < 8; ++r) {
      float m = vmax[r];
      m = fmaxf(m, __shfl_xor(m, 1, 32));
      m = fmaxf(m, __shfl_xor(m, 2, 32));
      m = fmaxf(m, __shfl_xor(m, 4, 32));
      m = fmaxf(m, __shfl_xor(m, 8, 32));
      vmax[r] = m;
    }
    if (l15 == 0) {
      #pragma unroll
      for (int r = 0; r < 8; ++r)
        ldsMax2[msub * 16 + hi16 * 8 + r][nsplit] = vmax[r];
    }
    __syncthreads();
    rowmax = fmaxf(ldsMax2[msub * 16 + l15][0], ldsMax2[msub * 16 + l15][1]);
  }

  // ======================= Pass B: P = exp(S-max); context += P @ enc ========
  v8f C[16];
  {
    const v8f z = {0.f,0.f,0.f,0.f,0.f,0.f,0.f,0.f};
    #pragma unroll
    for (int t = 0; t < 16; ++t) C[t] = z;
  }
  float psum = 0.f;

  #pragma unroll 1
  for (int c0 = 0; c0 < IN_LEN / NSPLIT; c0 += CHUNK) {
    const int n0 = nBase + c0;
    __syncthreads();
    // stage enc chunk transposed: [h][n] f16, padded stride 80B (16B aligned)
    #pragma unroll
    for (int c = 0; c < 16; ++c) {
      const int row = c * 2 + (gt >> 6);
      const int hq  = gt & 63;
      float4 v = *(const float4*)(encB + (size_t)(n0 + row) * HDIM + hq * 4);
      char* tb = buf + (hq * 4) * TSTRIDE + row * 2;
      *(_Float16*)(tb)               = (_Float16)v.x;
      *(_Float16*)(tb + TSTRIDE)     = (_Float16)v.y;
      *(_Float16*)(tb + 2 * TSTRIDE) = (_Float16)v.z;
      *(_Float16*)(tb + 3 * TSTRIDE) = (_Float16)v.w;
    }
    __syncthreads();

    // A fragment of P (16x32, K = n): re-read own raw scores (L2-hot), exp.
    Frag pf;
    {
      const float* rr = attnB + (size_t)(m0 + l15) * IN_LEN + n0 + hi16 * 8;
      float4 a0 = *(const float4*)(rr);
      float4 a1 = *(const float4*)(rr + 4);
      float4 b0 = *(const float4*)(rr + 16);
      float4 b1 = *(const float4*)(rr + 20);
      float e0 = __expf(a0.x - rowmax), e1 = __expf(a0.y - rowmax);
      float e2 = __expf(a0.z - rowmax), e3 = __expf(a0.w - rowmax);
      float e4 = __expf(a1.x - rowmax), e5 = __expf(a1.y - rowmax);
      float e6 = __expf(a1.z - rowmax), e7 = __expf(a1.w - rowmax);
      float f0 = __expf(b0.x - rowmax), f1 = __expf(b0.y - rowmax);
      float f2 = __expf(b0.z - rowmax), f3 = __expf(b0.w - rowmax);
      float f4 = __expf(b1.x - rowmax), f5 = __expf(b1.y - rowmax);
      float f6 = __expf(b1.z - rowmax), f7 = __expf(b1.w - rowmax);
      psum += ((e0+e1)+(e2+e3)) + ((e4+e5)+(e6+e7))
            + ((f0+f1)+(f2+f3)) + ((f4+f5)+(f6+f7));
      pf.u[0] = pk2(e0,e1); pf.u[1] = pk2(e2,e3);
      pf.u[2] = pk2(e4,e5); pf.u[3] = pk2(e6,e7);
      pf.u[4] = pk2(f0,f1); pf.u[5] = pk2(f2,f3);
      pf.u[6] = pk2(f4,f5); pf.u[7] = pk2(f6,f7);
    }

    #pragma unroll
    for (int t = 0; t < 16; ++t) {   // 16 h-tiles of the 256-wide context
      Frag bf;
      const char* tb = buf + (t * 16 + l15) * TSTRIDE + hi16 * 16;
      uint4 lo = *(const uint4*)(tb);
      uint4 hi = *(const uint4*)(tb + 32);
      bf.u[0]=lo.x; bf.u[1]=lo.y; bf.u[2]=lo.z; bf.u[3]=lo.w;
      bf.u[4]=hi.x; bf.u[5]=hi.y; bf.u[6]=hi.z; bf.u[7]=hi.w;
      C[t] = __builtin_amdgcn_wmma_f32_16x16x32_f16(
                 false, pf.v, false, bf.v, (short)0, C[t], false, false);
    }
  }

  // row sums: lanes l and l+16 hold the two halves of row l15
  psum += __shfl_xor(psum, 16, 32);
  if (lane < 16) ldsSum2[msub * 16 + lane][nsplit] = psum;
  __syncthreads();

  // combine partial context across the two n-splits (reuse staging LDS)
  #pragma unroll
  for (int phase = 0; phase < 2; ++phase) {
    if (nsplit == 1 && (msub >> 1) == phase) {
      float* dst = (float*)(ldsBuf + (msub & 1) * 16384);
      #pragma unroll
      for (int t = 0; t < 16; ++t)
        #pragma unroll
        for (int r = 0; r < 8; ++r)
          dst[(t * 8 + r) * 32 + lane] = C[t][r];
    }
    __syncthreads();
    if (nsplit == 0 && (msub >> 1) == phase) {
      const float* src = (const float*)(ldsBuf + (msub & 1) * 16384);
      #pragma unroll
      for (int t = 0; t < 16; ++t)
        #pragma unroll
        for (int r = 0; r < 8; ++r)
          C[t][r] += src[(t * 8 + r) * 32 + lane];
    }
    __syncthreads();
  }

  // scale by 1/rowsum and write context
  if (nsplit == 0) {
    float rinv[8];
    #pragma unroll
    for (int r = 0; r < 8; ++r) {
      const int lr = msub * 16 + hi16 * 8 + r;
      rinv[r] = 1.0f / (ldsSum2[lr][0] + ldsSum2[lr][1]);
    }
    float* crow = ctx + (size_t)(b * OUT_LEN + m0 + hi16 * 8) * HDIM + l15;
    #pragma unroll
    for (int t = 0; t < 16; ++t)
      #pragma unroll
      for (int r = 0; r < 8; ++r)
        crow[(size_t)r * HDIM + t * 16] = C[t][r] * rinv[r];
  }

  // ======================= Pass C: finalize attn = exp(S-max)/sum ============
  #pragma unroll 1
  for (int r16 = 0; r16 < 16; ++r16) {
    const int   lr   = msub * 16 + r16;
    const float mrow = fmaxf(ldsMax2[lr][0], ldsMax2[lr][1]);
    const float isr  = 1.0f / (ldsSum2[lr][0] + ldsSum2[lr][1]);
    float* row = attnB + (size_t)(m0 + r16) * IN_LEN + nBase;
    #pragma unroll 1
    for (int c = 0; c < (IN_LEN / NSPLIT) / 128; ++c) {
      float4* p = (float4*)(row + c * 128 + lane * 4);
      float4 v = *p;
      v.x = __expf(v.x - mrow) * isr;
      v.y = __expf(v.y - mrow) * isr;
      v.z = __expf(v.z - mrow) * isr;
      v.w = __expf(v.w - mrow) * isr;
      *p = v;
    }
  }
}

extern "C" void kernel_launch(void* const* d_in, const int* in_sizes, int n_in,
                              void* d_out, int out_size, void* d_ws, size_t ws_size,
                              hipStream_t stream) {
  const float* q    = (const float*)d_in[0];
  const float* enc  = (const float*)d_in[1];
  const int*   lens = (const int*)d_in[2];
  float* ctx  = (float*)d_out;
  float* attn = ctx + (size_t)NB * OUT_LEN * HDIM;   // context first, then attn
  dim3 grid(NB * (OUT_LEN / M_TILE));                // 128 workgroups
  attn_fused_kernel<<<grid, dim3(256), 0, stream>>>(q, enc, lens, ctx, attn);
  (void)in_sizes; (void)n_in; (void)out_size; (void)d_ws; (void)ws_size;
}
